// Attention_66511863546359
// MI455X (gfx1250) — matrix operand order repaired and checked by reading, not compiled, not run on hardware
//
#include <hip/hip_runtime.h>
#include <hip/hip_bf16.h>

typedef __attribute__((ext_vector_type(16))) __bf16 v16bf;
typedef __attribute__((ext_vector_type(8)))  __bf16 v8bf;
typedef __attribute__((ext_vector_type(8)))  float  v8f;
typedef unsigned int u32x4 __attribute__((ext_vector_type(4)));
typedef int          i32x8 __attribute__((ext_vector_type(8)));
typedef int          i32x4 __attribute__((ext_vector_type(4)));

#define DIM   256
#define SEQ   4096
#define HEADS 8
#define DH    32
#define BATCH 2

__device__ __forceinline__ v8f wmma_bf16(v16bf a, v16bf b, v8f c) {
    // D = A(16x32) * B(32x16) + C, f32 accumulate
    return __builtin_amdgcn_wmma_f32_16x16x32_bf16(
        /*neg_a=*/false, a, /*neg_b=*/false, b,
        /*c_mod=*/(short)0, c, /*reuse_a=*/false, /*reuse_b=*/false);
}

__device__ __forceinline__ unsigned lds_byte_off(const void* p) {
    // low 32 bits of a generic pointer into LDS = wave-relative LDS byte offset
    return (unsigned)(unsigned long long)(uintptr_t)p;
}

// TDM: DMA a 2-D bf16 tile (32 rows x 32 elements) from global into LDS.
// tensor_d0 = contiguous row length (elements), stride0 = row stride (elements),
// tensor_d1 = number of rows in the tensor. Tile is 32x32 fixed.
__device__ __forceinline__ void tdm_load_tile32x32(unsigned lds_off, const void* gaddr,
                                                   unsigned tensor_d0, unsigned tensor_d1,
                                                   unsigned stride0) {
    unsigned long long ga = (unsigned long long)(uintptr_t)gaddr;
    u32x4 g0;
    g0[0] = 1u;                                        // count=1, user descriptor
    g0[1] = lds_off;                                   // lds_addr (bytes)
    g0[2] = (unsigned)(ga & 0xFFFFFFFFu);              // global_addr[31:0]
    g0[3] = (unsigned)((ga >> 32) & 0x01FFFFFFu)       // global_addr[56:32]
          | (2u << 30);                                // type = 2 ("image")
    i32x8 g1;
    g1[0] = (int)(1u << 16);                           // wg_mask=0, data_size=1 (2 bytes)
    g1[1] = (int)(tensor_d0 << 16);                    // tensor_dim0[15:0]
    g1[2] = (int)((tensor_d0 >> 16) | (tensor_d1 << 16)); // dim0 hi | tensor_dim1 lo
    g1[3] = (int)((tensor_d1 >> 16) | (32u << 16));    // dim1 hi | tile_dim0 = 32
    g1[4] = 32;                                        // tile_dim1 = 32, tile_dim2 = 0
    g1[5] = (int)stride0;                              // tensor_dim0_stride[31:0]
    g1[6] = 0;                                         // stride hi | dim1_stride lo
    g1[7] = 0;
    i32x4 z4 = {0, 0, 0, 0};
#if defined(__clang_major__) && (__clang_major__ >= 23)
    i32x8 z8 = {0, 0, 0, 0, 0, 0, 0, 0};
    __builtin_amdgcn_tensor_load_to_lds(g0, g1, z4, z4, z8, 0);
#else
    __builtin_amdgcn_tensor_load_to_lds(g0, g1, z4, z4, 0);
#endif
}

// ---------------------------------------------------------------------------
// Kernel 1: qkv = x @ W_qkv + b_qkv, scattered to bf16
//   Q,K : [B*H, SEQ, DH] (token-major) ;  V : [B*H, DH, SEQ] (dim-major = V^T)
// One wave computes one 16x16 tile of the 8192x768 output.
// ---------------------------------------------------------------------------
__global__ void qkv_kernel(const float* __restrict__ x,
                           const float* __restrict__ Wqkv,
                           const float* __restrict__ bqkv,
                           __bf16* __restrict__ Qb,
                           __bf16* __restrict__ Kb,
                           __bf16* __restrict__ Vtb) {
    const int lane = threadIdx.x & 31;
    const int wave = threadIdx.x >> 5;
    const int tile = blockIdx.x * 8 + wave;          // 512*48 tiles
    const int tn = tile % 48, tm = tile / 48;
    const int row0 = tm * 16, col0 = tn * 16;
    const int half = lane >> 4, l16 = lane & 15;
    const int arow = row0 + l16;                      // A-frag row for lane
    const int bcol = col0 + l16;                      // B-frag column for lane

    v8f acc = {};
    for (int k0 = 0; k0 < DIM; k0 += 32) {
        // A fragment: dims k0 + half*16 + {0..15}
        const float* ap = x + (size_t)arow * DIM + k0 + half * 16;
        float4 a0 = *reinterpret_cast<const float4*>(ap + 0);
        float4 a1 = *reinterpret_cast<const float4*>(ap + 4);
        float4 a2 = *reinterpret_cast<const float4*>(ap + 8);
        float4 a3 = *reinterpret_cast<const float4*>(ap + 12);
        v16bf a;
        a[0]=(__bf16)a0.x; a[1]=(__bf16)a0.y; a[2]=(__bf16)a0.z; a[3]=(__bf16)a0.w;
        a[4]=(__bf16)a1.x; a[5]=(__bf16)a1.y; a[6]=(__bf16)a1.z; a[7]=(__bf16)a1.w;
        a[8]=(__bf16)a2.x; a[9]=(__bf16)a2.y; a[10]=(__bf16)a2.z; a[11]=(__bf16)a2.w;
        a[12]=(__bf16)a3.x; a[13]=(__bf16)a3.y; a[14]=(__bf16)a3.z; a[15]=(__bf16)a3.w;
        // B fragment: K rows k0 + half*16 + e at column bcol
        const float* bp = Wqkv + (size_t)(k0 + half * 16) * 768 + bcol;
        v16bf b;
        #pragma unroll
        for (int e = 0; e < 16; ++e) b[e] = (__bf16)bp[(size_t)e * 768];
        acc = wmma_bf16(a, b, acc);
    }

    // Epilogue: bias add + scatter. kind/head are uniform per 16-col tile.
    const float bias = bqkv[bcol];
    const int kind = bcol / 256;            // 0=q 1=k 2=v
    const int hh   = (bcol % 256) / DH;
    const int dd   = bcol % DH;
    if (kind == 2) {
        // V^T layout: [bh][dd][t]
        #pragma unroll
        for (int r = 0; r < 8; ++r) {
            int i = row0 + half * 8 + r;
            int bb = i >> 12, t = i & 4095;
            Vtb[(((size_t)(bb * HEADS + hh)) * DH + dd) * SEQ + t] = (__bf16)(acc[r] + bias);
        }
    } else {
        __bf16* dst = (kind == 0) ? Qb : Kb;
        #pragma unroll
        for (int r = 0; r < 8; ++r) {
            int i = row0 + half * 8 + r;
            int bb = i >> 12, t = i & 4095;
            dst[(((size_t)(bb * HEADS + hh)) * SEQ + t) * DH + dd] = (__bf16)(acc[r] + bias);
        }
    }
}

// ---------------------------------------------------------------------------
// Kernel 2: flash attention. 8 waves per block share one (b,h); each wave owns
// 16 query rows. K / V^T key blocks (32 tokens) are TDM-staged into
// double-buffered LDS; wave 0 drives the TDM pipeline with TENSORcnt.
// ---------------------------------------------------------------------------
__global__ void attn_kernel(const __bf16* __restrict__ Qb,
                            const __bf16* __restrict__ Kb,
                            const __bf16* __restrict__ Vtb,
                            __bf16* __restrict__ Ob) {
    __shared__ __bf16 Kt[2][32 * 32];   // [buf][token][dim]  2 KB each
    __shared__ __bf16 Vt[2][32 * 32];   // [buf][dim][token]  2 KB each
    __shared__ __bf16 Pl[8][16 * 32];   // per-wave P tile

    const int lane = threadIdx.x & 31;
    const int wave = threadIdx.x >> 5;
    const int bh = blockIdx.x >> 5;                     // 16 (b,h) pairs
    const int qt = (blockIdx.x & 31) * 8 + wave;        // query tile 0..255
    const int half = lane >> 4, l16 = lane & 15;

    const __bf16* Qp  = Qb  + ((size_t)bh * SEQ + qt * 16) * DH;
    const __bf16* Kp  = Kb  + (size_t)bh * SEQ * DH;
    const __bf16* Vtp = Vtb + (size_t)bh * DH * SEQ;

    // Q A-fragment: row l16, dims half*16 + {0..15}
    v16bf qa;
    {
        v8bf lo = *reinterpret_cast<const v8bf*>(Qp + l16 * DH + half * 16);
        v8bf hi = *reinterpret_cast<const v8bf*>(Qp + l16 * DH + half * 16 + 8);
        #pragma unroll
        for (int e = 0; e < 8; ++e) { qa[e] = lo[e]; qa[8 + e] = hi[e]; }
    }

    v8f acc0 = {}, acc1 = {};
    float m[8], lsum[8];
    #pragma unroll
    for (int r = 0; r < 8; ++r) { m[r] = -3.0e38f; lsum[r] = 0.0f; }

    __bf16* P = Pl[wave];
    const float scale = 0.0625f;            // DIM^-0.5 = 1/16 (full-dim scaling!)

    // Prime the TDM pipeline: stage block 0 into buffer 0.
    if (wave == 0) {
        tdm_load_tile32x32(lds_byte_off(&Kt[0][0]), Kp,  /*d0=*/DH,  /*d1=*/SEQ, /*stride=*/DH);
        tdm_load_tile32x32(lds_byte_off(&Vt[0][0]), Vtp, /*d0=*/SEQ, /*d1=*/DH,  /*stride=*/SEQ);
    }

    for (int kb = 0; kb < SEQ; kb += 32) {
        const int cur = (kb >> 5) & 1;
        if (wave == 0) {
            if (kb + 32 < SEQ) {
                const int nxt = cur ^ 1;
                tdm_load_tile32x32(lds_byte_off(&Kt[nxt][0]),
                                   Kp + (size_t)(kb + 32) * DH, DH, SEQ, DH);
                tdm_load_tile32x32(lds_byte_off(&Vt[nxt][0]),
                                   Vtp + (size_t)(kb + 32), SEQ, DH, SEQ);
                __builtin_amdgcn_s_wait_tensorcnt(2);   // current block landed
            } else {
                __builtin_amdgcn_s_wait_tensorcnt(0);
            }
        }
        __syncthreads();                                // LDS block visible to all

        const __bf16* kt = Kt[cur];
        const __bf16* vt = Vt[cur];

        // K B-fragments from LDS: row = key token, 16 contiguous dims per lane.
        v16bf kf0, kf1;
        {
            const __bf16* kp0 = kt + l16 * 32 + half * 16;
            const __bf16* kp1 = kt + (16 + l16) * 32 + half * 16;
            v8bf a = *reinterpret_cast<const v8bf*>(kp0);
            v8bf b = *reinterpret_cast<const v8bf*>(kp0 + 8);
            v8bf c = *reinterpret_cast<const v8bf*>(kp1);
            v8bf d = *reinterpret_cast<const v8bf*>(kp1 + 8);
            #pragma unroll
            for (int e = 0; e < 8; ++e) {
                kf0[e] = a[e]; kf0[8 + e] = b[e];
                kf1[e] = c[e]; kf1[8 + e] = d[e];
            }
        }
        v8f z = {};
        v8f s0 = wmma_bf16(qa, kf0, z);     // scores vs keys kb..kb+15
        v8f s1 = wmma_bf16(qa, kf1, z);     // scores vs keys kb+16..kb+31

        // Online softmax (row r lives in this lane-half; butterfly over 16 cols).
        float p0[8], p1[8];
        #pragma unroll
        for (int r = 0; r < 8; ++r) {
            float a0 = s0[r] * scale, a1 = s1[r] * scale;
            float mx = fmaxf(a0, a1);
            mx = fmaxf(mx, __shfl_xor(mx, 1));
            mx = fmaxf(mx, __shfl_xor(mx, 2));
            mx = fmaxf(mx, __shfl_xor(mx, 4));
            mx = fmaxf(mx, __shfl_xor(mx, 8));
            float nm = fmaxf(m[r], mx);
            float corr = __expf(m[r] - nm);
            float e0 = __expf(a0 - nm), e1 = __expf(a1 - nm);
            float ps = e0 + e1;
            ps += __shfl_xor(ps, 1);
            ps += __shfl_xor(ps, 2);
            ps += __shfl_xor(ps, 4);
            ps += __shfl_xor(ps, 8);
            lsum[r] = lsum[r] * corr + ps;
            m[r] = nm;
            acc0[r] *= corr; acc1[r] *= corr;
            p0[r] = e0; p1[r] = e1;
        }

        // C-layout -> A-layout transpose of P through per-wave LDS.
        #pragma unroll
        for (int r = 0; r < 8; ++r) {
            P[(half * 8 + r) * 32 + l16]      = (__bf16)p0[r];
            P[(half * 8 + r) * 32 + 16 + l16] = (__bf16)p1[r];
        }
        __builtin_amdgcn_wave_barrier();          // per-wave DS ops are in-order
        asm volatile("" ::: "memory");

        v16bf pa;                                 // P A-fragment: row l16, K = keys
        {
            v8bf lo = *reinterpret_cast<const v8bf*>(P + l16 * 32 + half * 16);
            v8bf hi = *reinterpret_cast<const v8bf*>(P + l16 * 32 + half * 16 + 8);
            #pragma unroll
            for (int e = 0; e < 8; ++e) { pa[e] = lo[e]; pa[8 + e] = hi[e]; }
        }

        // V B-fragments from transposed LDS tile: row = dim, 16 contiguous tokens.
        v16bf vf0, vf1;
        {
            const __bf16* vp0 = vt + l16 * 32 + half * 16;          // dims 0..15
            const __bf16* vp1 = vt + (16 + l16) * 32 + half * 16;   // dims 16..31
            v8bf a = *reinterpret_cast<const v8bf*>(vp0);
            v8bf b = *reinterpret_cast<const v8bf*>(vp0 + 8);
            v8bf c = *reinterpret_cast<const v8bf*>(vp1);
            v8bf d = *reinterpret_cast<const v8bf*>(vp1 + 8);
            #pragma unroll
            for (int e = 0; e < 8; ++e) {
                vf0[e] = a[e]; vf0[8 + e] = b[e];
                vf1[e] = c[e]; vf1[8 + e] = d[e];
            }
        }
        acc0 = wmma_bf16(pa, vf0, acc0);
        acc1 = wmma_bf16(pa, vf1, acc1);

        __syncthreads();    // everyone done with buffer `cur` before it is reloaded
    }

    // Normalize and store to token-major bf16 [B*SEQ, DIM] for the proj GEMM.
    const int b = bh >> 3, h = bh & 7;
    #pragma unroll
    for (int r = 0; r < 8; ++r) {
        float inv = 1.0f / lsum[r];
        int t = qt * 16 + half * 8 + r;
        size_t rowbase = ((size_t)b * SEQ + t) * DIM + h * DH;
        Ob[rowbase + l16]      = (__bf16)(acc0[r] * inv);
        Ob[rowbase + 16 + l16] = (__bf16)(acc1[r] * inv);
    }
}

// ---------------------------------------------------------------------------
// Kernel 3: out = attn_out @ W_proj + b_proj   (8192x256 @ 256x256, fp32 out)
// ---------------------------------------------------------------------------
__global__ void proj_kernel(const __bf16* __restrict__ Ob,
                            const float* __restrict__ Wp,
                            const float* __restrict__ bp,
                            float* __restrict__ out) {
    const int lane = threadIdx.x & 31;
    const int wave = threadIdx.x >> 5;
    const int tile = blockIdx.x * 8 + wave;      // 512*16 tiles
    const int tn = tile & 15, tm = tile >> 4;
    const int row0 = tm * 16, col0 = tn * 16;
    const int half = lane >> 4, l16 = lane & 15;
    const int arow = row0 + l16;
    const int bcol = col0 + l16;

    v8f acc = {};
    for (int k0 = 0; k0 < DIM; k0 += 32) {
        const __bf16* ap = Ob + (size_t)arow * DIM + k0 + half * 16;
        v8bf lo = *reinterpret_cast<const v8bf*>(ap);
        v8bf hi = *reinterpret_cast<const v8bf*>(ap + 8);
        v16bf a;
        #pragma unroll
        for (int e = 0; e < 8; ++e) { a[e] = lo[e]; a[8 + e] = hi[e]; }
        const float* wp = Wp + (size_t)(k0 + half * 16) * DIM + bcol;
        v16bf b;
        #pragma unroll
        for (int e = 0; e < 16; ++e) b[e] = (__bf16)wp[(size_t)e * DIM];
        acc = wmma_bf16(a, b, acc);
    }

    const float bias = bp[bcol];
    #pragma unroll
    for (int r = 0; r < 8; ++r) {
        int i = row0 + half * 8 + r;
        out[(size_t)i * DIM + bcol] = acc[r] + bias;
    }
}

// ---------------------------------------------------------------------------
extern "C" void kernel_launch(void* const* d_in, const int* in_sizes, int n_in,
                              void* d_out, int out_size, void* d_ws, size_t ws_size,
                              hipStream_t stream) {
    const float* x     = (const float*)d_in[0];
    const float* Wqkv  = (const float*)d_in[1];
    const float* bqkv  = (const float*)d_in[2];
    const float* Wproj = (const float*)d_in[3];
    const float* bproj = (const float*)d_in[4];
    float* out = (float*)d_out;

    char* ws = (char*)d_ws;
    const size_t qkv_bytes = (size_t)BATCH * HEADS * SEQ * DH * sizeof(__bf16); // 4 MB each
    __bf16* Qb  = (__bf16*)(ws);
    __bf16* Kb  = (__bf16*)(ws + qkv_bytes);
    __bf16* Vtb = (__bf16*)(ws + 2 * qkv_bytes);       // [B*H, DH, SEQ]
    __bf16* Ob  = (__bf16*)(ws + 3 * qkv_bytes);       // [B*SEQ, DIM]

    // 8192x768 output, 16x16 tile per wave, 8 waves per 256-thread block.
    qkv_kernel<<<dim3((512 * 48) / 8), dim3(256), 0, stream>>>(x, Wqkv, bqkv, Qb, Kb, Vtb);
    // 16 (b,h) x 32 blocks; 8 query tiles per block.
    attn_kernel<<<dim3(512), dim3(256), 0, stream>>>(Qb, Kb, Vtb, Ob);
    // 8192x256 output -> 8192 tiles
    proj_kernel<<<dim3((512 * 16) / 8), dim3(256), 0, stream>>>(Ob, Wproj, bproj, out);
}
